// MultiHeadSelfAttention_83382495084952
// MI455X (gfx1250) — matrix operand order repaired and checked by reading, not compiled, not run on hardware
//
#include <hip/hip_runtime.h>
#include <hip/hip_bf16.h>
#include <math.h>

// ---------------------------------------------------------------------------
// MI455X / gfx1250 multi-head self-attention, bf16 WMMA path.
//   B=4, S=2048, D=1024, H=16, Dk=64
// Pipeline:
//   (1) weights fp32 -> bf16, transposed + swizzled into WMMA B-operand layout
//   (2) Q/K/V projections: 32x32-per-wave WMMA GEMM, custom epilogues
//       (Q epilogue folds 1/sqrt(dk)*log2(e) so attention uses exp2 directly)
//   (3) flash attention: 32-row query tile per wave (2x K/V reuse),
//       online softmax in the exp2 domain; row-sums accumulated via a
//       ones-column WMMA (no sum shuffles); max reduction batched stage-wise
//   (4) output projection -> fp32 d_out
// ---------------------------------------------------------------------------

typedef __bf16 bf16;
typedef __attribute__((ext_vector_type(16))) __bf16 v16bf;
typedef __attribute__((ext_vector_type(8)))  __bf16 v8bf;
typedef __attribute__((ext_vector_type(8)))  float  v8f;

static constexpr int BATCH = 4;
static constexpr int S     = 2048;
static constexpr int D     = 1024;
static constexpr int H     = 16;
static constexpr int DK    = 64;
static constexpr int BS    = BATCH * S;   // 8192 rows

// 1/sqrt(64) * log2(e): folded into the Q projection output.
#define QSCALE 0.18033688011112042f

union Af { v16bf v; v8bf h[2]; __bf16 e[16]; };

#define WMMA_BF16(A, B, C) \
  __builtin_amdgcn_wmma_f32_16x16x32_bf16(false, (A), false, (B), (short)0, (C), false, false)

// Swizzled B-operand storage for a logical [Ktot x Ntot] bf16 matrix, tiled
// 32(K) x 16(N). Within a tile, element order is exactly the per-lane WMMA
// B layout: lane = ((k%32 / 8)&1)*16 + (n%16); element s holds K = (s&7) + 8*(lane/16) + 16*(s/8).
__device__ __forceinline__ size_t bswz(int k, int n, int ntiles /*Ntot/16*/) {
  int tk = k >> 5, kk = k & 31;
  int tn = n >> 4, nn = n & 15;
  int lane = (((kk >> 3) & 1) << 4) | nn;
  int j = ((kk & 7) | ((kk >> 4) << 3)) >> 1, half = kk & 1;
  return (((size_t)(tk * ntiles + tn)) << 9) + (size_t)(lane << 4) + (size_t)(j << 1) + (size_t)half;
}

// ---------------------------------------------------------------------------
// Kernel 1: fp32 weight [e][d] (torch y = x @ W^T) -> bf16 swizzled B layout
// of logical Wt[k=d][n=e]. Writes are the swizzled linear order (coalesced).
// ---------------------------------------------------------------------------
__global__ __launch_bounds__(256) void swizzle_weight(const float* __restrict__ W,
                                                      bf16* __restrict__ out) {
  int o = blockIdx.x * 256 + threadIdx.x;          // o in [0, D*D)
  int tile = o >> 9, within = o & 511;
  int lane = within >> 4, rem = within & 15;
  int j = rem >> 1, half = rem & 1;
  const int ntiles = D >> 4;                        // 64 N-tiles per K-row
  int tk = tile / ntiles, tn = tile - tk * ntiles;
  int k = (tk << 5) + ((j >> 2) << 4) + ((lane >> 4) << 3) + ((j & 3) << 1) + half;  // d index
  int n = (tn << 4) + (lane & 15);                           // e index
  out[o] = (bf16)W[(size_t)n * D + k];              // Wt[k][n] = W[n][k]
}

// ---------------------------------------------------------------------------
// Kernel 2: WMMA GEMM, Y[M=8192][N=1024] = A[M][K=1024] * Bsw + bias.
// One wave computes a 32x32 tile (2x2 C tiles, 4 WMMAs per K-step of 32).
// MODE 0: A=f32 (q), write bf16 Qh[b][h][s][dk], scaled by QSCALE
// MODE 1: A=f32 (k), write bf16 swizzled B layout of Kh^T [dk][s] per (b,h)
// MODE 2: A=f32 (v), write bf16 swizzled B layout of Vh   [s][dk] per (b,h)
// MODE 3: A=bf16 (concat), write fp32 flat [M][N] (final output)
// ---------------------------------------------------------------------------
template <int MODE>
__global__ __launch_bounds__(256) void gemm32x32(const void* __restrict__ Aptr,
                                                 const bf16* __restrict__ Bsw,
                                                 const float* __restrict__ bias,
                                                 void* __restrict__ Out) {
  const int N = D, K = D;
  int lane = threadIdx.x & 31;
  int wave = threadIdx.x >> 5;
  int hi = lane >> 4, lo = lane & 15;
  int tileId = blockIdx.x * 8 + wave;               // 8192 tiles total
  const int tilesN = N >> 5;                        // 32
  int tm = tileId / tilesN, tn = tileId - tm * tilesN;
  int m0 = tm << 5, n0 = tn << 5;

  v8f acc[2][2] = {};

  for (int k0 = 0; k0 < K; k0 += 32) {
    Af a[2];
    if constexpr (MODE == 3) {
      const bf16* A = (const bf16*)Aptr;
#pragma unroll
      for (int mt = 0; mt < 2; ++mt) {
        const bf16* p = A + (size_t)(m0 + mt * 16 + lo) * K + k0 + 8 * hi;
        a[mt].h[0] = *(const v8bf*)p;
        a[mt].h[1] = *(const v8bf*)(p + 16);
      }
    } else {
      const float* A = (const float*)Aptr;
#pragma unroll
      for (int mt = 0; mt < 2; ++mt) {
        const float* p = A + (size_t)(m0 + mt * 16 + lo) * K + k0 + 8 * hi;
        float4 x0 = *(const float4*)(p);
        float4 x1 = *(const float4*)(p + 4);
        float4 y0 = *(const float4*)(p + 16);
        float4 y1 = *(const float4*)(p + 20);
        a[mt].e[0]  = (bf16)x0.x; a[mt].e[1]  = (bf16)x0.y;
        a[mt].e[2]  = (bf16)x0.z; a[mt].e[3]  = (bf16)x0.w;
        a[mt].e[4]  = (bf16)x1.x; a[mt].e[5]  = (bf16)x1.y;
        a[mt].e[6]  = (bf16)x1.z; a[mt].e[7]  = (bf16)x1.w;
        a[mt].e[8]  = (bf16)y0.x; a[mt].e[9]  = (bf16)y0.y;
        a[mt].e[10] = (bf16)y0.z; a[mt].e[11] = (bf16)y0.w;
        a[mt].e[12] = (bf16)y1.x; a[mt].e[13] = (bf16)y1.y;
        a[mt].e[14] = (bf16)y1.z; a[mt].e[15] = (bf16)y1.w;
      }
    }
    Af bm[2];
#pragma unroll
    for (int nt = 0; nt < 2; ++nt) {
      size_t tb = (((size_t)((k0 >> 5) * (N >> 4) + (n0 >> 4) + nt)) << 9) + (size_t)(lane << 4);
      bm[nt].h[0] = *(const v8bf*)(Bsw + tb);
      bm[nt].h[1] = *(const v8bf*)(Bsw + tb + 8);
    }
#pragma unroll
    for (int mt = 0; mt < 2; ++mt)
#pragma unroll
      for (int nt = 0; nt < 2; ++nt)
        acc[mt][nt] = WMMA_BF16(a[mt].v, bm[nt].v, acc[mt][nt]);
  }

  // Epilogue: bias add + layout-specific store.
#pragma unroll
  for (int nt = 0; nt < 2; ++nt) {
    float bv = bias[n0 + nt * 16 + lo];
#pragma unroll
    for (int mt = 0; mt < 2; ++mt) {
#pragma unroll
      for (int r = 0; r < 8; ++r) {
        int m = m0 + mt * 16 + r + 8 * hi;           // C layout: row r + 8*(lane>=16)
        int n = n0 + nt * 16 + lo;                   //           col = lane & 15
        float val = acc[mt][nt][r] + bv;
        if constexpr (MODE == 0) {
          int b = m >> 11, s = m & 2047, hh = n >> 6, dk = n & 63;
          ((bf16*)Out)[(((size_t)(b * H + hh) * S + s) << 6) + dk] = (bf16)(val * QSCALE);
        } else if constexpr (MODE == 1) {
          int b = m >> 11, t = m & 2047, hh = n >> 6, dk = n & 63;
          size_t base = (size_t)(b * H + hh) * (size_t)(DK * S);
          ((bf16*)Out)[base + bswz(dk, t, S >> 4)] = (bf16)val;   // Kh^T [dk][s]
        } else if constexpr (MODE == 2) {
          int b = m >> 11, t = m & 2047, hh = n >> 6, dk = n & 63;
          size_t base = (size_t)(b * H + hh) * (size_t)(S * DK);
          ((bf16*)Out)[base + bswz(t, dk, DK >> 4)] = (bf16)val;  // Vh [s][dk]
        } else {
          ((float*)Out)[(size_t)m * N + n] = val;
        }
      }
    }
  }
}

// ---------------------------------------------------------------------------
// Kernel 3: flash attention. One wave owns a 32-row query tile of one (b,h).
// Per 32-key chunk: 8 WMMAs for S = Q K^T, online softmax in the exp2 domain
// (scale pre-folded into Qh). The softmax denominator is accumulated with a
// ones-column WMMA (Oacc[4] = alpha*Oacc[4] + P*ones), so NO cross-lane sum
// shuffles are needed; only the row-max uses ds_bpermute, batched stage-wise
// (4 waits/chunk). P re-layout via 2KB per-wave LDS; 10 WMMAs for O/l update.
// ---------------------------------------------------------------------------
__global__ __launch_bounds__(256) void attn_flash(const bf16* __restrict__ Qh,
                                                  const bf16* __restrict__ Ksw,
                                                  const bf16* __restrict__ Vsw,
                                                  bf16* __restrict__ concat) {
  __shared__ __align__(16) bf16 pbuf[8][32 * 32];   // per-wave P tile (bf16)

  int lane = threadIdx.x & 31;
  int wave = threadIdx.x >> 5;
  int hi = lane >> 4, lo = lane & 15;
  int wid = blockIdx.x * 8 + wave;                  // [0, 4096)
  int stile = wid & (S / 32 - 1);                   // 64 query tiles
  int h = (wid >> 6) & (H - 1);
  int b = wid >> 10;
  int s0 = stile << 5;

  // Q A-operand tiles: 2 row-tiles x 2 K-halves (16x32 each), held in regs.
  Af Aq[2][2];
#pragma unroll
  for (int mt = 0; mt < 2; ++mt) {
    const bf16* qbase =
        Qh + (((size_t)((b * H + h) * S + s0 + mt * 16 + lo)) << 6) + 8 * hi;
    Aq[mt][0].h[0] = *(const v8bf*)(qbase);
    Aq[mt][0].h[1] = *(const v8bf*)(qbase + 16);
    Aq[mt][1].h[0] = *(const v8bf*)(qbase + 32);
    Aq[mt][1].h[1] = *(const v8bf*)(qbase + 48);
  }

  // All-ones B operand for the row-sum WMMA.
  Af ones;
#pragma unroll
  for (int i = 0; i < 16; ++i) ones.e[i] = (bf16)1.0f;

  const bf16* kb = Ksw + (size_t)(b * H + h) * (size_t)(DK * S);
  const bf16* vb = Vsw + (size_t)(b * H + h) * (size_t)(S * DK);

  float m_i[2][8];
  v8f Oacc[2][5] = {};                              // [4] = running row-sum l_i
#pragma unroll
  for (int mt = 0; mt < 2; ++mt)
#pragma unroll
    for (int r = 0; r < 8; ++r) m_i[mt][r] = -1e30f;

  for (int t0 = 0; t0 < S; t0 += 32) {
    // ---- prefetch next chunk's K and V tiles into the cache hierarchy ----
    if (t0 + 32 < S) {
      int tn_next = (t0 + 32) >> 4;
      __builtin_prefetch(kb + ((((size_t)(0 * (S >> 4) + tn_next)) << 9) + (lane << 4)), 0, 1);
      __builtin_prefetch(kb + ((((size_t)(1 * (S >> 4) + tn_next)) << 9) + (lane << 4)), 0, 1);
      __builtin_prefetch(vb + ((((size_t)(((t0 + 32) >> 5) * (DK >> 4))) << 9) + (lane << 4)), 0, 1);
    }

    // ---- S = Q K^T for 32 keys (two 16-col subtiles, both row-tiles) ----
    Af Bk[2][2];
#pragma unroll
    for (int kh = 0; kh < 2; ++kh)
#pragma unroll
      for (int ts = 0; ts < 2; ++ts) {
        size_t tb = (((size_t)(kh * (S >> 4) + (t0 >> 4) + ts)) << 9) + (size_t)(lane << 4);
        Bk[kh][ts].h[0] = *(const v8bf*)(kb + tb);
        Bk[kh][ts].h[1] = *(const v8bf*)(kb + tb + 8);
      }
    v8f Sacc[2][2];
#pragma unroll
    for (int mt = 0; mt < 2; ++mt)
#pragma unroll
      for (int ts = 0; ts < 2; ++ts) {
        v8f z = {};
        z = WMMA_BF16(Aq[mt][0].v, Bk[0][ts].v, z);
        z = WMMA_BF16(Aq[mt][1].v, Bk[1][ts].v, z);
        Sacc[mt][ts] = z;
      }

    // ---- row max: local, then 4 batched shuffle stages over all 16 rows ----
    float mx[2][8];
#pragma unroll
    for (int mt = 0; mt < 2; ++mt)
#pragma unroll
      for (int r = 0; r < 8; ++r)
        mx[mt][r] = fmaxf(Sacc[mt][0][r], Sacc[mt][1][r]);
#pragma unroll
    for (int st = 1; st <= 8; st <<= 1) {
      float t[2][8];
#pragma unroll
      for (int mt = 0; mt < 2; ++mt)
#pragma unroll
        for (int r = 0; r < 8; ++r) t[mt][r] = __shfl_xor(mx[mt][r], st, 32);
#pragma unroll
      for (int mt = 0; mt < 2; ++mt)
#pragma unroll
        for (int r = 0; r < 8; ++r) mx[mt][r] = fmaxf(mx[mt][r], t[mt][r]);
    }

    // ---- exp2-domain softmax numerators; rescale O and l accumulators ----
    bf16* pl = pbuf[wave];
#pragma unroll
    for (int mt = 0; mt < 2; ++mt) {
#pragma unroll
      for (int r = 0; r < 8; ++r) {
        float mnew  = fmaxf(m_i[mt][r], mx[mt][r]);
        float alpha = exp2f(m_i[mt][r] - mnew);
        float p0 = exp2f(Sacc[mt][0][r] - mnew);
        float p1 = exp2f(Sacc[mt][1][r] - mnew);
        m_i[mt][r] = mnew;
#pragma unroll
        for (int j = 0; j < 5; ++j) Oacc[mt][j][r] *= alpha;
        int row = mt * 16 + r + 8 * hi;             // C layout -> LDS row-major
        pl[row * 32 + lo]      = (bf16)p0;
        pl[row * 32 + 16 + lo] = (bf16)p1;
      }
    }

    // ---- reload P as WMMA A operands (row-major LDS, k0 = 0) ----
    Af Ap[2];
#pragma unroll
    for (int mt = 0; mt < 2; ++mt) {
      const bf16* pa = pl + (mt * 16 + lo) * 32 + 8 * hi;
      Ap[mt].h[0] = *(const v8bf*)(pa);
      Ap[mt].h[1] = *(const v8bf*)(pa + 16);
    }

    // ---- O += P V ; l += P * ones (all on the matrix pipe) ----
#pragma unroll
    for (int j = 0; j < 4; ++j) {
      size_t tb = (((size_t)((t0 >> 5) * (DK >> 4) + j)) << 9) + (size_t)(lane << 4);
      Af Bv;
      Bv.h[0] = *(const v8bf*)(vb + tb);
      Bv.h[1] = *(const v8bf*)(vb + tb + 8);
#pragma unroll
      for (int mt = 0; mt < 2; ++mt)
        Oacc[mt][j] = WMMA_BF16(Ap[mt].v, Bv.v, Oacc[mt][j]);
    }
#pragma unroll
    for (int mt = 0; mt < 2; ++mt)
      Oacc[mt][4] = WMMA_BF16(Ap[mt].v, ones.v, Oacc[mt][4]);
  }

  // ---- normalize (each lane holds its row's l in Oacc[4]) and write ----
#pragma unroll
  for (int mt = 0; mt < 2; ++mt)
#pragma unroll
    for (int r = 0; r < 8; ++r) {
      float linv = 1.0f / Oacc[mt][4][r];
#pragma unroll
      for (int j = 0; j < 4; ++j) {
        float val = Oacc[mt][j][r] * linv;
        int s = s0 + mt * 16 + r + 8 * hi;
        int col = h * DK + j * 16 + lo;
        concat[((size_t)b * S + s) * D + col] = (bf16)val;
      }
    }
}

// ---------------------------------------------------------------------------
// Host-side launch
// ---------------------------------------------------------------------------
extern "C" void kernel_launch(void* const* d_in, const int* in_sizes, int n_in,
                              void* d_out, int out_size, void* d_ws, size_t ws_size,
                              hipStream_t stream) {
  (void)in_sizes; (void)n_in; (void)out_size; (void)ws_size;

  const float* q    = (const float*)d_in[0];
  const float* k    = (const float*)d_in[1];
  const float* v    = (const float*)d_in[2];
  const float* Wq_w = (const float*)d_in[3];
  const float* Wq_b = (const float*)d_in[4];
  const float* Wk_w = (const float*)d_in[5];
  const float* Wk_b = (const float*)d_in[6];
  const float* Wv_w = (const float*)d_in[7];
  const float* Wv_b = (const float*)d_in[8];
  const float* W0_w = (const float*)d_in[9];
  const float* W0_b = (const float*)d_in[10];

  // Workspace layout (bf16 elements): 4x weights (1M each) + 4x 8M tensors = 72MB
  bf16* Wq_sw = (bf16*)d_ws;
  bf16* Wk_sw = Wq_sw + (size_t)D * D;
  bf16* Wv_sw = Wk_sw + (size_t)D * D;
  bf16* W0_sw = Wv_sw + (size_t)D * D;
  bf16* Qh    = W0_sw + (size_t)D * D;
  bf16* Ksw   = Qh  + (size_t)BATCH * H * S * DK;
  bf16* Vsw   = Ksw + (size_t)BATCH * H * S * DK;
  bf16* Cc    = Vsw + (size_t)BATCH * H * S * DK;

  dim3 blk(256);
  // (1) weight conversion + swizzle
  swizzle_weight<<<(D * D) / 256, blk, 0, stream>>>(Wq_w, Wq_sw);
  swizzle_weight<<<(D * D) / 256, blk, 0, stream>>>(Wk_w, Wk_sw);
  swizzle_weight<<<(D * D) / 256, blk, 0, stream>>>(Wv_w, Wv_sw);
  swizzle_weight<<<(D * D) / 256, blk, 0, stream>>>(W0_w, W0_sw);

  // (2) projections: 8192 wave-tiles / 8 waves per block = 1024 blocks
  gemm32x32<0><<<1024, blk, 0, stream>>>(q, Wq_sw, Wq_b, Qh);
  gemm32x32<1><<<1024, blk, 0, stream>>>(k, Wk_sw, Wk_b, Ksw);
  gemm32x32<2><<<1024, blk, 0, stream>>>(v, Wv_sw, Wv_b, Vsw);

  // (3) flash attention: B*H*(S/32) = 4096 waves -> 512 blocks
  attn_flash<<<512, blk, 0, stream>>>(Qh, Ksw, Vsw, Cc);

  // (4) output projection -> fp32 d_out
  gemm32x32<3><<<1024, blk, 0, stream>>>(Cc, W0_sw, W0_b, (float*)d_out);
}
